// IntelligentNeuron_44976897523902
// MI455X (gfx1250) — matrix operand order repaired and checked by reading, not compiled, not run on hardware
//
#include <hip/hip_runtime.h>

// ---------------- problem constants ----------------
constexpr int L    = 2048;   // sequence length
constexpr int DM   = 1024;   // d_model
constexpr int DI   = 2048;   // d_inner
constexpr int DS   = 16;     // d_state
constexpr int RK   = 64;     // dt_rank
constexpr int XPD  = RK + 2 * DS;  // 96

// ---------------- types ----------------
typedef __bf16 bf16_t;
typedef bf16_t v8bf  __attribute__((ext_vector_type(8)));
typedef bf16_t v16bf __attribute__((ext_vector_type(16)));
typedef float  v8f   __attribute__((ext_vector_type(8)));

__device__ __forceinline__ bf16_t f2bf(float f) { return (bf16_t)f; }

// A-fragment load per ISA 16-bit A layout: two 16B chunks (K groups of 8)
__device__ __forceinline__ v16bf ldA(const bf16_t* p) {
    v8bf lo = *(const v8bf*)(p);
    v8bf hi = *(const v8bf*)(p + 16);
    return __builtin_shufflevector(lo, hi,
             0,1,2,3,4,5,6,7,8,9,10,11,12,13,14,15);
}

__device__ __forceinline__ v8f wmma_bf16(v16bf a, v16bf b, v8f c) {
    return __builtin_amdgcn_wmma_f32_16x16x32_bf16(
        /*neg_a=*/false, a, /*neg_b=*/false, b,
        /*c_mod=*/(short)0, c, /*reuse_a=*/false, /*reuse_b=*/false);
}

// ---------------- cast kernels ----------------
__global__ void cast_f32_to_bf16(const float* __restrict__ src,
                                 bf16_t* __restrict__ dst, int n) {
    int i = blockIdx.x * blockDim.x + threadIdx.x;
    if (i < n) dst[i] = f2bf(src[i]);
}

// dt = x_dbl[:, 0:64] -> bf16, packed [L, RK]
__global__ void cast_dt(const float* __restrict__ xdbl,
                        bf16_t* __restrict__ dtb) {
    int i = blockIdx.x * blockDim.x + threadIdx.x;   // over L*RK
    int l = i >> 6, r = i & 63;
    dtb[i] = f2bf(xdbl[(size_t)l * XPD + r]);
}

// ---------------- WMMA GEMM:  C[M,N] = A[M,K] @ B[N,K]^T  ----------------
// 32x32 macro-tile per wave (2x2 grid of 16x16 WMMA tiles) for 2x register
// reuse of both operands. 8 waves / block.
// epilogue: 0 = plain f32 store; 1 = softplus(acc + bias[col]) (dt_proj)
__global__ __launch_bounds__(256) void gemm_bf16_wmma(
    const bf16_t* __restrict__ A, const bf16_t* __restrict__ B,
    float* __restrict__ C, const float* __restrict__ bias,
    int N, int K, int lda, int ldb, int ldc, int epilogue)
{
    const int lane   = threadIdx.x & 31;
    const int wid    = threadIdx.x >> 5;
    const int tilesN = N >> 5;                    // 32-wide macro tiles
    const int tile   = blockIdx.x * 8 + wid;
    const int tm     = tile / tilesN;
    const int tn     = tile % tilesN;

    const int hiHalf = lane >> 4;                 // 0: lanes 0-15, 1: 16-31
    const int lm     = lane & 15;
    const int aOff   = hiHalf * 8;                // A K sub-offset (ISA layout)
    const int bOff   = hiHalf * 8;                // B K sub-offset (ISA layout)

    const bf16_t* __restrict__ Arow0 = A + (size_t)(tm * 32 + lm) * lda;
    const bf16_t* __restrict__ Arow1 = Arow0 + (size_t)16 * lda;
    const bf16_t* __restrict__ Bcol0 = B + (size_t)(tn * 32 + lm) * ldb;
    const bf16_t* __restrict__ Bcol1 = Bcol0 + (size_t)16 * ldb;

    v8f acc00 = {}, acc01 = {}, acc10 = {}, acc11 = {};
    for (int k0 = 0; k0 < K; k0 += 32) {
        v16bf a0 = ldA(Arow0 + k0 + aOff);
        v16bf a1 = ldA(Arow1 + k0 + aOff);
        v16bf b0 = ldA(Bcol0 + k0 + bOff);
        v16bf b1 = ldA(Bcol1 + k0 + bOff);
        acc00 = wmma_bf16(a0, b0, acc00);
        acc01 = wmma_bf16(a0, b1, acc01);
        acc10 = wmma_bf16(a1, b0, acc10);
        acc11 = wmma_bf16(a1, b1, acc11);
    }

    // C/D layout: VGPR r -> row (tileRow + hiHalf*8 + r), col = tileCol + lm
    v8f accs[4] = {acc00, acc01, acc10, acc11};
    #pragma unroll
    for (int i = 0; i < 2; i++) {
        #pragma unroll
        for (int j = 0; j < 2; j++) {
            const v8f acc  = accs[i * 2 + j];
            const int rb   = tm * 32 + i * 16 + hiHalf * 8;
            const int col  = tn * 32 + j * 16 + lm;
            #pragma unroll
            for (int r = 0; r < 8; r++) {
                float v = acc[r];
                if (epilogue == 1) {
                    v += bias[col];
                    v = (v > 20.f) ? v : log1pf(__expf(v));   // softplus
                }
                C[(size_t)(rb + r) * ldc + col] = v;
            }
        }
    }
}

// ---------------- depthwise causal conv (width 4) + SiLU ----------------
__global__ void conv_silu_kernel(const float* __restrict__ xz,
                                 const float* __restrict__ cw,
                                 const float* __restrict__ cb,
                                 float* __restrict__ ucv,
                                 bf16_t* __restrict__ ucvb)
{
    int i = blockIdx.x * blockDim.x + threadIdx.x;   // over L*DI
    int c = i % DI, t = i / DI;
    float s = cb[c];
    #pragma unroll
    for (int k = 0; k < 4; k++) {
        int tt = t + k - 3;
        if (tt >= 0) s += cw[c * 4 + k] * xz[(size_t)tt * (2 * DI) + c];
    }
    float u = s / (1.f + __expf(-s));                // SiLU
    ucv[i]  = u;
    ucvb[i] = f2bf(u);
}

// ---------------- async global->LDS helpers (CDNA5 ASYNCcnt path) -------
__device__ __forceinline__ void async_g2l_b128(void* lds_ptr, const void* gptr) {
    unsigned lds_off = (unsigned)(uintptr_t)lds_ptr;      // LDS aperture: low 32b
    unsigned long long ga = (unsigned long long)(uintptr_t)gptr;
    asm volatile("global_load_async_to_lds_b128 %0, %1, off"
                 :: "v"(lds_off), "v"(ga) : "memory");
}
__device__ __forceinline__ void wait_asynccnt0() {
    asm volatile("s_wait_asynccnt 0" ::: "memory");
}

// ---------------- selective scan ----------------
// One thread per channel; N=16 state in registers. B/C vectors (shared by all
// channels) double-buffered in LDS via GLOBAL_LOAD_ASYNC_TO_LDS_B128:
// next 64-timestep chunk streams in while the current one is consumed.
__global__ __launch_bounds__(256) void scan_kernel(
    const float* __restrict__ delta, const float* __restrict__ ucv,
    const float* __restrict__ xz,    const float* __restrict__ xdbl,
    const float* __restrict__ A_log, const float* __restrict__ Dp,
    bf16_t* __restrict__ yb)
{
    __shared__ float sBC[2][64][2 * DS];     // [buf][tt][B(16) | C(16)]
    const int tid = threadIdx.x;
    const int c   = blockIdx.x * 256 + tid;

    float Ac[DS], h[DS];
    #pragma unroll
    for (int n = 0; n < DS; n++) {
        Ac[n] = -__expf(A_log[(size_t)c * DS + n]);  // A = -exp(A_log)
        h[n]  = 0.f;
    }
    const float Dc = Dp[c];

    // stage a 64-step chunk: 2048 f32 = 256 thr x 2 x b128
    auto stage = [&](int buf, int tc) {
        #pragma unroll
        for (int half = 0; half < 2; half++) {
            int e  = half * 1024 + tid * 4;          // 4 consecutive f32
            int tt = e >> 5, n = e & 31;
            async_g2l_b128(&sBC[buf][tt][n],
                           &xdbl[(size_t)(tc + tt) * XPD + RK + n]);
        }
    };

    stage(0, 0);
    int pb = 0;
    for (int tc = 0; tc < L; tc += 64, pb ^= 1) {
        wait_asynccnt0();            // this wave's copies have landed
        __syncthreads();             // ... and therefore everyone's have
        if (tc + 64 < L) {
            stage(pb ^ 1, tc + 64);  // prefetch next chunk into other buffer
            __builtin_prefetch(&delta[(size_t)(tc + 64) * DI + c], 0, 0);
            __builtin_prefetch(&ucv  [(size_t)(tc + 64) * DI + c], 0, 0);
        }

        for (int tt = 0; tt < 64; tt++) {
            const int t = tc + tt;
            float dv = delta[(size_t)t * DI + c];
            float uv = ucv  [(size_t)t * DI + c];
            float du = dv * uv;
            float y  = 0.f;
            #pragma unroll
            for (int n = 0; n < DS; n++) {
                h[n] = __expf(dv * Ac[n]) * h[n] + du * sBC[pb][tt][n];
                y += h[n] * sBC[pb][tt][DS + n];
            }
            float zv = xz[(size_t)t * (2 * DI) + DI + c];
            y = (y + uv * Dc) * (zv / (1.f + __expf(-zv)));   // +D*u, *silu(z)
            yb[(size_t)t * DI + c] = f2bf(y);
        }
        __syncthreads();             // all reads of buf pb done before reuse
    }
}

// ---------------- host orchestration ----------------
extern "C" void kernel_launch(void* const* d_in, const int* in_sizes, int n_in,
                              void* d_out, int out_size, void* d_ws, size_t ws_size,
                              hipStream_t stream) {
    const float* x       = (const float*)d_in[0];   // [L, DM]
    const float* W_in    = (const float*)d_in[1];   // [2*DI, DM]
    const float* conv_w  = (const float*)d_in[2];   // [DI, 1, 4]
    const float* conv_b  = (const float*)d_in[3];   // [DI]
    const float* W_xproj = (const float*)d_in[4];   // [XPD, DI]
    const float* W_dt    = (const float*)d_in[5];   // [DI, RK]
    const float* b_dt    = (const float*)d_in[6];   // [DI]
    const float* A_log   = (const float*)d_in[7];   // [DI, DS]
    const float* D_param = (const float*)d_in[8];   // [DI]
    const float* W_out   = (const float*)d_in[9];   // [DM, DI]
    float* out = (float*)d_out;                     // [L, DM]

    // workspace layout (256B aligned slices)
    char* p = (char*)d_ws;
    auto take = [&](size_t bytes) {
        char* q = p;
        p += (bytes + 255) & ~size_t(255);
        return q;
    };
    bf16_t* xb      = (bf16_t*)take((size_t)L * DM * 2);
    bf16_t* W_inb   = (bf16_t*)take((size_t)2 * DI * DM * 2);
    float*  xz      = (float*) take((size_t)L * 2 * DI * 4);
    float*  ucv     = (float*) take((size_t)L * DI * 4);
    bf16_t* ucvb    = (bf16_t*)take((size_t)L * DI * 2);
    bf16_t* W_xprjb = (bf16_t*)take((size_t)XPD * DI * 2);
    float*  xdbl    = (float*) take((size_t)L * XPD * 4);
    bf16_t* dtb     = (bf16_t*)take((size_t)L * RK * 2);
    bf16_t* W_dtb   = (bf16_t*)take((size_t)DI * RK * 2);
    float*  delta   = (float*) take((size_t)L * DI * 4);
    bf16_t* ybuf    = (bf16_t*)take((size_t)L * DI * 2);
    bf16_t* W_outb  = (bf16_t*)take((size_t)DM * DI * 2);

    const int T = 256;
    auto gblocks = [](int M, int N) { return (M / 32) * (N / 32) / 8; };

    // casts of inputs to bf16
    cast_f32_to_bf16<<<(L * DM) / T, T, 0, stream>>>(x, xb, L * DM);
    cast_f32_to_bf16<<<(2 * DI * DM) / T, T, 0, stream>>>(W_in, W_inb, 2 * DI * DM);
    cast_f32_to_bf16<<<(XPD * DI) / T, T, 0, stream>>>(W_xproj, W_xprjb, XPD * DI);
    cast_f32_to_bf16<<<(DI * RK) / T, T, 0, stream>>>(W_dt, W_dtb, DI * RK);
    cast_f32_to_bf16<<<(DM * DI) / T, T, 0, stream>>>(W_out, W_outb, DM * DI);

    // GEMM1: xz[L, 2*DI] = x @ W_in^T      (M=2048, N=4096, K=1024)
    gemm_bf16_wmma<<<gblocks(L, 2 * DI), T, 0, stream>>>(
        xb, W_inb, xz, nullptr, 2 * DI, DM, DM, DM, 2 * DI, 0);

    // conv + SiLU -> ucv / ucvb
    conv_silu_kernel<<<(L * DI) / T, T, 0, stream>>>(xz, conv_w, conv_b, ucv, ucvb);

    // GEMM2: x_dbl[L, 96] = u @ W_xproj^T  (M=2048, N=96, K=2048)
    gemm_bf16_wmma<<<gblocks(L, XPD), T, 0, stream>>>(
        ucvb, W_xprjb, xdbl, nullptr, XPD, DI, DI, DI, XPD, 0);

    // dt slice -> bf16
    cast_dt<<<(L * RK) / T, T, 0, stream>>>(xdbl, dtb);

    // GEMM3: delta[L, DI] = softplus(dt @ W_dt^T + b_dt)  (M=2048, N=2048, K=64)
    gemm_bf16_wmma<<<gblocks(L, DI), T, 0, stream>>>(
        dtb, W_dtb, delta, b_dt, DI, RK, RK, RK, DI, 1);

    // selective scan (channel-parallel), async double-buffered B/C staging
    scan_kernel<<<DI / T, T, 0, stream>>>(delta, ucv, xz, xdbl, A_log, D_param, ybuf);

    // GEMM4: out[L, DM] = y @ W_out^T      (M=2048, N=1024, K=2048)
    gemm_bf16_wmma<<<gblocks(L, DM), T, 0, stream>>>(
        ybuf, W_outb, out, nullptr, DM, DI, DI, DI, DM, 0);

    (void)in_sizes; (void)n_in; (void)out_size; (void)ws_size;
}
